// LSTM_79559974191270
// MI455X (gfx1250) — compile-verified, matrix operand order checked by reading, not run on hardware
//
#include <hip/hip_runtime.h>
#include <hip/hip_bf16.h>
#include <math.h>

typedef unsigned short u16;
typedef __attribute__((ext_vector_type(16))) __bf16    v16bf;
typedef __attribute__((ext_vector_type(8)))  float     v8f;
typedef __attribute__((ext_vector_type(4)))  unsigned  uint4v;

union Frag { uint4v q[2]; v16bf m; };

constexpr int Tn   = 8192;   // timesteps
constexpr int Dn   = 2048;   // input dim (= K of precompute GEMM)
constexpr int Hn   = 2048;   // hidden dim
constexpr int G4   = 8192;   // 4*H gate columns
constexpr int HIDn = 4096;   // MLP hidden
constexpr int CSTR = 2056;   // padded LDS column stride (u16): 4112B -> 2-way max bank conflict

__device__ __forceinline__ u16 f2bf(float f) {
    unsigned u = __float_as_uint(f);
    unsigned r = u + 0x7fffu + ((u >> 16) & 1u);   // round-to-nearest-even
    return (u16)(r >> 16);
}
__device__ __forceinline__ float bfbits2f(unsigned hi16) { return __uint_as_float(hi16); }
__device__ __forceinline__ float sigmoidf_(float x) { return 1.0f / (1.0f + expf(-x)); }

// ---------------- conversion / init ----------------

__global__ void k_cvt(const float* __restrict__ src, u16* __restrict__ dst, size_t n) {
    for (size_t i = (size_t)blockIdx.x * blockDim.x + threadIdx.x; i < n;
         i += (size_t)gridDim.x * blockDim.x)
        dst[i] = f2bf(src[i]);
}

// src: R x C row-major fp32  ->  dst: C x R ("column major") bf16
__global__ void k_cvt_t(const float* __restrict__ src, u16* __restrict__ dst, int R, int C) {
    size_t n = (size_t)R * C;
    for (size_t i = (size_t)blockIdx.x * blockDim.x + threadIdx.x; i < n;
         i += (size_t)gridDim.x * blockDim.x) {
        size_t r = i / (size_t)C, c = i % (size_t)C;
        dst[c * (size_t)R + r] = f2bf(src[i]);
    }
}

__global__ void k_init(float* hbuf, unsigned* ctr) {
    int i = blockIdx.x * blockDim.x + threadIdx.x;
    if (i < Hn) hbuf[i] = 0.0f;
    if (i == 0) ctr[0] = 0u;
}

// ---------------- precompute GEMM: Xg = xseq @ Wi + b  (bf16 WMMA, pipelined) ----------------
// Abf: Tn x Dn row-major bf16 ; BT: G4 x Dn bf16 (transposed Wi, K contiguous)
// Block = 8 waves; wave tile 16(M) x 64(N); block tile 128 x 64.

__global__ __launch_bounds__(256) void k_gemm_xwi(
    const u16* __restrict__ Abf, const u16* __restrict__ BT,
    const float* __restrict__ bias, float* __restrict__ Xg)
{
    const int tid  = threadIdx.x;
    const int lane = tid & 31;
    const int wave = tid >> 5;
    const int half = lane >> 4;     // 0: lanes 0-15, 1: lanes 16-31
    const int lm   = lane & 15;
    const int row0 = blockIdx.y * 128 + wave * 16;
    const int col0 = blockIdx.x * 64;

    v8f acc[4] = {};

    const u16* aRow = Abf + (size_t)(row0 + lm) * Dn;
    const u16* bCol = BT  + (size_t)(col0 + lm) * Dn;

    // per ISA 16-bit A/B layout: lane holds K={0..7}+8*half (q0) and K={16..23}+8*half (q1)
    auto loadA = [&](Frag& f, int k0) {
        const u16* ap = aRow + k0 + half * 8;
        f.q[0] = *(const uint4v*)(ap);
        f.q[1] = *(const uint4v*)(ap + 16);
    };
    auto loadB = [&](Frag* f, int k0) {
#pragma unroll
        for (int tI = 0; tI < 4; ++tI) {
            const u16* bp = bCol + (size_t)(tI * 16) * Dn + k0 + half * 8;
            f[tI].q[0] = *(const uint4v*)(bp);
            f[tI].q[1] = *(const uint4v*)(bp + 16);
        }
    };

    Frag a0, a1;
    Frag b0[4], b1[4];
    loadA(a0, 0);
    loadB(b0, 0);

    for (int k0 = 0; k0 < Dn; k0 += 64) {
        // issue loads for k0+32 while computing on k0
        loadA(a1, k0 + 32);
        loadB(b1, k0 + 32);
#pragma unroll
        for (int tI = 0; tI < 4; ++tI)
            acc[tI] = __builtin_amdgcn_wmma_f32_16x16x32_bf16(
                false, a0.m, false, b0[tI].m, (short)0, acc[tI], false, false);

        const int k2 = (k0 + 64 < Dn) ? k0 + 64 : 0;   // last iter: harmless in-bounds reload
        loadA(a0, k2);
        loadB(b0, k2);
#pragma unroll
        for (int tI = 0; tI < 4; ++tI)
            acc[tI] = __builtin_amdgcn_wmma_f32_16x16x32_bf16(
                false, a1.m, false, b1[tI].m, (short)0, acc[tI], false, false);
    }

    // epilogue: C layout N = lane%16, M = vgpr + 8*half ; fuse bias
#pragma unroll
    for (int tI = 0; tI < 4; ++tI) {
        const int col = col0 + tI * 16 + lm;
        const float bb = bias[col];
#pragma unroll
        for (int r = 0; r < 8; ++r) {
            const int row = row0 + r + half * 8;
            Xg[(size_t)row * G4 + col] = acc[tI][r] + bb;
        }
    }
}

// ---------------- persistent sequential LSTM recurrence ----------------
// 128 blocks x 256 threads. Block owns 16 h-indices -> 64 gate columns, 4-way split-K.
// The block's 256KB slice of WhT is cached in LDS (320KB/WGP on CDNA5) across all 8192 steps.

__global__ __launch_bounds__(256) void k_lstm(
    const u16* __restrict__ WhT, const float* __restrict__ Xg,
    float* __restrict__ hbuf, unsigned* __restrict__ ctr)
{
    __shared__ __align__(16) u16 wsh[64 * CSTR];   // 263168 B weight slice
    __shared__ float hsh[Hn];                      // 8 KB hidden state broadcast
    __shared__ float red[256];

    const int tid    = threadIdx.x;
    const int nb     = gridDim.x;
    const int j0     = blockIdx.x * 16;
    const int colIdx = tid & 63;            // 0..63 : gate(2b) x local-j(4b)
    const int quad   = tid >> 6;            // split-K quarter
    const int col    = (colIdx >> 4) * Hn + j0 + (colIdx & 15);
    float creg = 0.0f;                      // cell state lives in registers (tid<16)

    // one-time: stage this thread's weight stripe into LDS (it only reads what it wrote)
    {
        const u16* gp = WhT + (size_t)col * Hn + quad * 512;
        u16* lp = wsh + colIdx * CSTR + quad * 512;
        for (int kk = 0; kk < 512; kk += 8)
            *(uint4v*)(lp + kk) = *(const uint4v*)(gp + kk);
    }
    const u16* wl = wsh + colIdx * CSTR + quad * 512;
    __syncthreads();

    for (int t = 0; t < Tn; ++t) {
        for (int i = tid; i < Hn; i += 256) hsh[i] = hbuf[i];
        __syncthreads();

        float acc = 0.0f;
        const float* hq = hsh + quad * 512;
#pragma unroll 4
        for (int kk = 0; kk < 512; kk += 8) {          // ds_load_b128: 8 bf16 per iter
            uint4v w = *(const uint4v*)(wl + kk);
#pragma unroll
            for (int e = 0; e < 4; ++e) {
                unsigned u = w[e];
                acc = fmaf(bfbits2f(u << 16),         hq[kk + 2 * e],     acc);
                acc = fmaf(bfbits2f(u & 0xffff0000u), hq[kk + 2 * e + 1], acc);
            }
        }
        red[tid] = acc;
        __syncthreads();
        if (tid < 64) {
            const int c2 = (tid >> 4) * Hn + j0 + (tid & 15);
            red[tid] = red[tid] + red[64 + tid] + red[128 + tid] + red[192 + tid]
                     + Xg[(size_t)t * G4 + c2];        // x@Wi + b precomputed
        }
        __syncthreads();
        if (tid < 16) {                                 // flax gate order i,f,g,o
            float gi = red[tid],      gf = red[16 + tid];
            float gg = red[32 + tid], go = red[48 + tid];
            float cn = sigmoidf_(gf) * creg + sigmoidf_(gi) * tanhf(gg);
            creg = cn;
            hbuf[j0 + tid] = sigmoidf_(go) * tanhf(cn);
        }
        __threadfence();
        __syncthreads();
        if (tid == 0) {                                 // device-wide step barrier
            __hip_atomic_fetch_add(ctr, 1u, __ATOMIC_RELEASE, __HIP_MEMORY_SCOPE_AGENT);
            const unsigned target = (unsigned)(t + 1) * (unsigned)nb;
            while (__hip_atomic_load(ctr, __ATOMIC_ACQUIRE, __HIP_MEMORY_SCOPE_AGENT) < target)
                __builtin_amdgcn_s_sleep(1);
        }
        __syncthreads();
        __threadfence();
    }
}

// ---------------- tiny fp32 MLP head ----------------

__global__ void k_fc_relu(const float* __restrict__ W, const float* __restrict__ bias,
                          const float* __restrict__ in, float* __restrict__ out, int K, int N)
{
    int col = blockIdx.x * blockDim.x + threadIdx.x;
    if (col >= N) return;
    float acc = 0.0f;
    for (int k = 0; k < K; ++k) acc = fmaf(in[k], W[(size_t)k * N + col], acc);
    out[col] = fmaxf(acc + bias[col], 0.0f);
}

__global__ void k_fc_out(const float* __restrict__ Wout, const float* __restrict__ bout,
                         const float* __restrict__ in, float* __restrict__ out, int K)
{
    __shared__ float red[256];
    int tid = threadIdx.x;
    float acc = 0.0f;
    for (int i = tid; i < K; i += 256) acc += in[i] * Wout[i];
    red[tid] = acc;
    __syncthreads();
    for (int s = 128; s > 0; s >>= 1) { if (tid < s) red[tid] += red[tid + s]; __syncthreads(); }
    if (tid == 0) out[0] = red[0] + bout[0];
}

// ---------------- launch ----------------

extern "C" void kernel_launch(void* const* d_in, const int* in_sizes, int n_in,
                              void* d_out, int out_size, void* d_ws, size_t ws_size,
                              hipStream_t stream)
{
    (void)in_sizes; (void)n_in; (void)out_size; (void)ws_size;
    const float* xseq = (const float*)d_in[0];
    const float* Wi   = (const float*)d_in[1];
    const float* Wh   = (const float*)d_in[2];
    const float* b    = (const float*)d_in[3];
    const float* W1   = (const float*)d_in[4];
    const float* b1   = (const float*)d_in[5];
    const float* W2   = (const float*)d_in[6];
    const float* b2   = (const float*)d_in[7];
    const float* Wout = (const float*)d_in[8];
    const float* bout = (const float*)d_in[9];
    float* out = (float*)d_out;

    char* ws = (char*)d_ws;
    size_t off = 0;
    float*    Xg   = (float*)(ws + off); off += (size_t)Tn * G4 * sizeof(float);  // 256 MB
    u16*      Abf  = (u16*)  (ws + off); off += (size_t)Tn * Dn * sizeof(u16);    // 32 MB
    u16*      WiT  = (u16*)  (ws + off); off += (size_t)G4 * Dn * sizeof(u16);    // 32 MB
    u16*      WhT  = (u16*)  (ws + off); off += (size_t)G4 * Hn * sizeof(u16);    // 32 MB
    float*    hbuf = (float*)(ws + off); off += 4096;
    unsigned* ctr  = (unsigned*)(ws + off); off += 256;
    float*    a1   = (float*)(ws + off); off += HIDn * sizeof(float);
    float*    a2   = (float*)(ws + off); off += HIDn * sizeof(float);

    k_init<<<(Hn + 255) / 256, 256, 0, stream>>>(hbuf, ctr);
    k_cvt  <<<2048, 256, 0, stream>>>(xseq, Abf, (size_t)Tn * Dn);
    k_cvt_t<<<2048, 256, 0, stream>>>(Wi, WiT, Dn, G4);
    k_cvt_t<<<2048, 256, 0, stream>>>(Wh, WhT, Hn, G4);

    dim3 gg(G4 / 64, Tn / 128);
    k_gemm_xwi<<<gg, 256, 0, stream>>>(Abf, WiT, b, Xg);

    k_lstm<<<Hn / 16, 256, 0, stream>>>(WhT, Xg, hbuf, ctr);

    k_fc_relu<<<HIDn / 256, 256, 0, stream>>>(W1, b1, hbuf, a1, Hn, HIDn);
    k_fc_relu<<<HIDn / 256, 256, 0, stream>>>(W2, b2, a1, a2, HIDn, HIDn);
    k_fc_out <<<1, 256, 0, stream>>>(Wout, bout, a2, out, HIDn);
}
